// UnetFormer_85839216378421
// MI455X (gfx1250) — compile-verified, matrix operand order checked by reading, not run on hardware
//
#include <hip/hip_runtime.h>
#include <stdint.h>

// ---------------------------------------------------------------------------
// CDNA5 (gfx1250) UnetFormer forward. wave32, WMMA bf16 16x16x32.
// ---------------------------------------------------------------------------

typedef __attribute__((ext_vector_type(16))) __bf16 v16bf;
typedef __attribute__((ext_vector_type(8)))  float  v8f;

union FragU { uint32_t u[8]; v16bf v; };

#define PI2F 9.869604401089358f

__device__ __forceinline__ uint16_t f2bf(float x) {
  uint32_t u = __float_as_uint(x);
  uint32_t r = u + 0x7FFFu + ((u >> 16) & 1u);   // round-to-nearest-even
  return (uint16_t)(r >> 16);
}
__device__ __forceinline__ float leaky(float x) { return x >= 0.f ? x : 0.2f * x; }
__device__ __forceinline__ float gelu_tanh(float x) {
  float x3 = x * x * x;
  float t = tanhf(0.7978845608f * (x + 0.044715f * x3));
  return 0.5f * x * (1.f + t);
}

// ---------------------------------------------------------------------------
// Stage A: patch extraction + FFT-equivalent blur (separable circular conv)
// grid = 8192 blocks, 512 threads (one voxel per thread)
// ---------------------------------------------------------------------------
__global__ __launch_bounds__(512)
void patch_blur_kernel(const float* __restrict__ img,
                       const float* __restrict__ bfac,
                       float* __restrict__ outp)
{
  __shared__ float sa[512];
  __shared__ float sb[512];
  int s = blockIdx.x;
  int b = s >> 9, pi = s & 511;
  int Z = pi >> 6, Y = (pi >> 3) & 7, X = pi & 7;
  int v = threadIdx.x;                 // 0..511
  int z = v >> 6, y = (v >> 3) & 7, x = v & 7;

  sa[v] = img[(((size_t)(b * 64 + Z * 8 + z)) * 64 + Y * 8 + y) * 64 + X * 8 + x];

  // h[n] = (1/8) sum_k exp(-c k^2) cos(2 pi k n / 8),   c = b * pi^2 / 16
  float c  = bfac[s] * (PI2F / 16.0f);
  float m1 = __expf(-c), m2 = __expf(-4.f * c), m3 = __expf(-9.f * c), m4 = __expf(-16.f * c);
  const float r2 = 0.70710678118f;
  float h[8];
  h[0] = (1.f + 2.f * m1 + 2.f * m2 + 2.f * m3 + m4) * 0.125f;
  h[1] = (1.f + 2.f * m1 * r2 - 2.f * m3 * r2 - m4) * 0.125f;
  h[2] = (1.f - 2.f * m2 + m4) * 0.125f;
  h[3] = (1.f - 2.f * m1 * r2 + 2.f * m3 * r2 - m4) * 0.125f;
  h[4] = (1.f - 2.f * m1 + 2.f * m2 - 2.f * m3 + m4) * 0.125f;
  h[5] = h[3]; h[6] = h[2]; h[7] = h[1];
  __syncthreads();

  float acc = 0.f;                     // x axis
#pragma unroll
  for (int t = 0; t < 8; ++t) acc += h[t] * sa[(v & ~7) | ((x - t) & 7)];
  sb[v] = acc; __syncthreads();

  acc = 0.f;                           // y axis
#pragma unroll
  for (int t = 0; t < 8; ++t) acc += h[t] * sb[(z << 6) | (((y - t) & 7) << 3) | x];
  sa[v] = acc; __syncthreads();

  acc = 0.f;                           // z axis
#pragma unroll
  for (int t = 0; t < 8; ++t) acc += h[t] * sa[((((z - t) & 7)) << 6) | (y << 3) | x];
  outp[(size_t)s * 512 + v] = acc;
}

// ---------------------------------------------------------------------------
// Stage B: fully fused conv encoder per subvolume (+ LN(u) + pos-emb add)
// grid = 8192 blocks, 256 threads
// ---------------------------------------------------------------------------
__global__ __launch_bounds__(256)
void encoder_kernel(const float* __restrict__ P,
                    const float* __restrict__ w0, const float* __restrict__ b0c,
                    const float* __restrict__ g1, const float* __restrict__ be1,
                    const float* __restrict__ w1, const float* __restrict__ b1c,
                    const float* __restrict__ g2, const float* __restrict__ be2,
                    const float* __restrict__ w2, const float* __restrict__ b2c,
                    const float* __restrict__ g3, const float* __restrict__ be3,
                    const float* __restrict__ wu, const float* __restrict__ bu,
                    const float* __restrict__ posln,
                    const float* __restrict__ lndg, const float* __restrict__ lndb,
                    float* __restrict__ xout)
{
  __shared__ float sIn[512], sW0[2048], sD0[2048], sD1[512], sD2[128], sU[256];
  __shared__ float sRA[256], sRB[256];
  int s = blockIdx.x; int pi = s & 511;
  int tid = threadIdx.x;

  for (int j = tid; j < 512;  j += 256) sIn[j] = P[(size_t)s * 512 + j];
  for (int j = tid; j < 2048; j += 256) sW0[j] = w0[j];
  __syncthreads();

  // conv0: 1ch 8^3 -> 32ch 4^3 (k4 s2 p1) + leaky
  for (int o = tid; o < 2048; o += 256) {
    int c = o >> 6, r = o & 63;
    int oz = r >> 4, oy = (r >> 2) & 3, ox = r & 3;
    float acc = b0c[c];
    for (int tz = 0; tz < 4; ++tz) { int iz = 2 * oz - 1 + tz; if ((unsigned)iz >= 8u) continue;
      for (int ty = 0; ty < 4; ++ty) { int iy = 2 * oy - 1 + ty; if ((unsigned)iy >= 8u) continue;
        for (int tx = 0; tx < 4; ++tx) { int ix = 2 * ox - 1 + tx; if ((unsigned)ix >= 8u) continue;
          acc += sW0[c * 64 + tz * 16 + ty * 4 + tx] * sIn[iz * 64 + iy * 8 + ix];
        } } }
    sD0[o] = leaky(acc);
  }
  __syncthreads();

  // GroupNorm G=4 over 32ch x 4^3 (512 vals/group)
  {
    int g = tid >> 6, lt = tid & 63;
    float s1 = 0.f, s2 = 0.f;
    for (int j = 0; j < 8; ++j) { float v = sD0[g * 512 + lt * 8 + j]; s1 += v; s2 += v * v; }
    sRA[tid] = s1; sRB[tid] = s2; __syncthreads();
    for (int off = 32; off > 0; off >>= 1) {
      if (lt < off) { sRA[tid] += sRA[tid + off]; sRB[tid] += sRB[tid + off]; }
      __syncthreads();
    }
    float mean = sRA[g << 6] * (1.f / 512.f);
    float var  = sRB[g << 6] * (1.f / 512.f) - mean * mean;
    float rs = rsqrtf(var + 1e-5f);
    for (int j = 0; j < 8; ++j) {
      int li = lt * 8 + j; int c = (g << 3) + (li >> 6);
      float v = sD0[g * 512 + li];
      sD0[g * 512 + li] = (v - mean) * rs * g1[c] + be1[c];
    }
  }
  __syncthreads();

  // conv1: 32ch 4^3 -> 64ch 2^3 + leaky  (weights from L2)
  for (int o = tid; o < 512; o += 256) {
    int c = o >> 3, r = o & 7;
    int oz = (r >> 2) & 1, oy = (r >> 1) & 1, ox = r & 1;
    float acc = b1c[c];
    const float* wrow = w1 + (size_t)c * 32 * 64;
    for (int ic = 0; ic < 32; ++ic) {
      const float* wi = wrow + ic * 64;
      const float* xi = sD0 + ic * 64;
      for (int tz = 0; tz < 4; ++tz) { int iz = 2 * oz - 1 + tz; if ((unsigned)iz >= 4u) continue;
        for (int ty = 0; ty < 4; ++ty) { int iy = 2 * oy - 1 + ty; if ((unsigned)iy >= 4u) continue;
          for (int tx = 0; tx < 4; ++tx) { int ix = 2 * ox - 1 + tx; if ((unsigned)ix >= 4u) continue;
            acc += wi[tz * 16 + ty * 4 + tx] * xi[iz * 16 + iy * 4 + ix];
          } } }
    }
    sD1[o] = leaky(acc);
  }
  __syncthreads();

  // GroupNorm G=2 over 64ch x 2^3 (256 vals/group)
  {
    int g = tid >> 7, lt = tid & 127;
    float s1 = 0.f, s2 = 0.f;
    for (int j = 0; j < 2; ++j) { float v = sD1[g * 256 + lt * 2 + j]; s1 += v; s2 += v * v; }
    sRA[tid] = s1; sRB[tid] = s2; __syncthreads();
    for (int off = 64; off > 0; off >>= 1) {
      if (lt < off) { sRA[tid] += sRA[tid + off]; sRB[tid] += sRB[tid + off]; }
      __syncthreads();
    }
    float mean = sRA[g << 7] * (1.f / 256.f);
    float var  = sRB[g << 7] * (1.f / 256.f) - mean * mean;
    float rs = rsqrtf(var + 1e-5f);
    for (int j = 0; j < 2; ++j) {
      int li = lt * 2 + j; int c = (g << 5) + (li >> 3);
      float v = sD1[g * 256 + li];
      sD1[g * 256 + li] = (v - mean) * rs * g2[c] + be2[c];
    }
  }
  __syncthreads();

  // conv2: 64ch 2^3 -> 128ch 1^3 + leaky (only taps 1..2 are in-range)
  float myd2 = 0.f;
  if (tid < 128) {
    int c = tid;
    float acc = b2c[c];
    const float* wrow = w2 + (size_t)c * 64 * 64;
    for (int ic = 0; ic < 64; ++ic) {
      const float* wi = wrow + ic * 64;
      const float* xi = sD1 + ic * 8;
      for (int tz = 1; tz < 3; ++tz)
        for (int ty = 1; ty < 3; ++ty)
          for (int tx = 1; tx < 3; ++tx)
            acc += wi[tz * 16 + ty * 4 + tx] * xi[(tz - 1) * 4 + (ty - 1) * 2 + (tx - 1)];
    }
    myd2 = leaky(acc);
    sD2[c] = myd2;
  }
  __syncthreads();

  // GroupNorm G=1 over 128 vals
  {
    float v = (tid < 128) ? myd2 : 0.f;
    sRA[tid] = v; sRB[tid] = v * v; __syncthreads();
    for (int off = 128; off > 0; off >>= 1) {
      if (tid < off) { sRA[tid] += sRA[tid + off]; sRB[tid] += sRB[tid + off]; }
      __syncthreads();
    }
    float mean = sRA[0] * (1.f / 128.f);
    float var  = sRB[0] * (1.f / 128.f) - mean * mean;
    float rs = rsqrtf(var + 1e-5f);
    __syncthreads();
    if (tid < 128) sD2[tid] = (myd2 - mean) * rs * g3[tid] + be3[tid];
  }
  __syncthreads();

  // wu: 1x1 conv 128 -> 256
  {
    float acc = bu[tid];
    const float* wrow = wu + (size_t)tid * 128;
    for (int ic = 0; ic < 128; ++ic) acc += wrow[ic] * sD2[ic];
    sU[tid] = acc;
  }
  __syncthreads();

  // LN(u) + posln[pi]
  {
    float v = sU[tid];
    sRA[tid] = v; __syncthreads();
    for (int off = 128; off > 0; off >>= 1) { if (tid < off) sRA[tid] += sRA[tid + off]; __syncthreads(); }
    float mean = sRA[0] * (1.f / 256.f);
    __syncthreads();
    float d = v - mean;
    sRA[tid] = d * d; __syncthreads();
    for (int off = 128; off > 0; off >>= 1) { if (tid < off) sRA[tid] += sRA[tid + off]; __syncthreads(); }
    float var = sRA[0] * (1.f / 256.f);
    float o = d * rsqrtf(var + 1e-5f) * lndg[tid] + lndb[tid];
    xout[(size_t)s * 256 + tid] = o + posln[(size_t)pi * 256 + tid];
  }
}

// ---------------------------------------------------------------------------
// LayerNorm over last dim = 256. One block (256 threads) per row.
// outF / outB optional (f32 / bf16).
// ---------------------------------------------------------------------------
__global__ __launch_bounds__(256)
void ln256_kernel(const float* __restrict__ in, const float* __restrict__ g,
                  const float* __restrict__ bia, float* __restrict__ outF,
                  uint16_t* __restrict__ outB)
{
  __shared__ float red[256];
  int row = blockIdx.x, t = threadIdx.x;
  size_t base = (size_t)row * 256;
  float v = in[base + t];
  red[t] = v; __syncthreads();
  for (int off = 128; off > 0; off >>= 1) { if (t < off) red[t] += red[t + off]; __syncthreads(); }
  float mean = red[0] * (1.f / 256.f);
  __syncthreads();
  float d = v - mean;
  red[t] = d * d; __syncthreads();
  for (int off = 128; off > 0; off >>= 1) { if (t < off) red[t] += red[t + off]; __syncthreads(); }
  float var = red[0] * (1.f / 256.f);
  float o = d * rsqrtf(var + 1e-5f) * g[t] + bia[t];
  if (outF) outF[base + t] = o;
  if (outB) outB[base + t] = f2bf(o);
}

// ---------------------------------------------------------------------------
// Softmax over rows of 512 (scores for one batch, all 8 heads), writes bf16.
// grid = (512 rows, 8 heads), 256 threads (2 elems each)
// ---------------------------------------------------------------------------
__global__ __launch_bounds__(256)
void softmax512_kernel(const float* __restrict__ S, uint16_t* __restrict__ Sb, float scale)
{
  __shared__ float red[256];
  int h = blockIdx.y, row = blockIdx.x, t = threadIdx.x;
  const float* r = S + ((size_t)h * 512 + row) * 512;
  float a = r[t] * scale, b = r[t + 256] * scale;
  red[t] = fmaxf(a, b); __syncthreads();
  for (int off = 128; off > 0; off >>= 1) { if (t < off) red[t] = fmaxf(red[t], red[t + off]); __syncthreads(); }
  float mx = red[0];
  __syncthreads();
  float ea = __expf(a - mx), eb = __expf(b - mx);
  red[t] = ea + eb; __syncthreads();
  for (int off = 128; off > 0; off >>= 1) { if (t < off) red[t] += red[t + off]; __syncthreads(); }
  float inv = 1.f / red[0];
  uint16_t* o = Sb + ((size_t)h * 512 + row) * 512;
  o[t] = f2bf(ea * inv); o[t + 256] = f2bf(eb * inv);
}

// ---------------------------------------------------------------------------
// f32 (L,K,N) -> bf16 transposed (L,N,K)
// ---------------------------------------------------------------------------
__global__ __launch_bounds__(256)
void conv_w_T(const float* __restrict__ W, uint16_t* __restrict__ out, int L, int K, int N)
{
  int idx = blockIdx.x * 256 + threadIdx.x;
  int total = L * K * N;
  if (idx >= total) return;
  int l = idx / (K * N); int rem = idx - l * K * N; int k = rem / N; int n = rem - k * N;
  out[((size_t)l * N + n) * K + k] = f2bf(W[idx]);
}

// ---------------------------------------------------------------------------
// WMMA GEMM: C[M x N] = A[M x K] * Bt[N x K]^T  (bf16 x bf16 -> f32)
// 8 waves/block; wave w -> 16x16 tile (mTile = blockIdx.y*8+w, nTile = blockIdx.x)
// modes: 0 = f32 out (+bias, +z*sCz)    1 = bf16 out (+bias, +z*sCz)
//        2 = f32 residual add (+bias)   3 = bf16 gelu(out+bias)
//        4 = bf16 per-head-transposed V write  vt[((b*8+h)*32+d)*512+tok]
// ---------------------------------------------------------------------------
__global__ __launch_bounds__(256)
void gemm_bf16_wmma(const uint16_t* __restrict__ A, int lda, long long sAz,
                    const uint16_t* __restrict__ Bt, int ldb, long long sBz,
                    const float* __restrict__ bias,
                    float* __restrict__ outF, uint16_t* __restrict__ outB,
                    int ldc, long long sCz,
                    int M, int K, int mode)
{
  int z = blockIdx.z;
  A  += (size_t)z * sAz;
  Bt += (size_t)z * sBz;
  int lane = threadIdx.x & 31, wave = threadIdx.x >> 5;
  int mBase = (blockIdx.y * 8 + wave) * 16;
  int nBase = blockIdx.x * 16;
  if (mBase >= M) return;
  int half = lane >> 4;                        // 0 or 1
  int rA = mBase + (lane & 15);
  int rB = nBase + (lane & 15);

  v8f acc = {};
  for (int k0 = 0; k0 < K; k0 += 32) {
    FragU fa, fb;
    const uint32_t* pa0 = (const uint32_t*)(A  + (size_t)rA * lda + k0 + half * 8);
    const uint32_t* pa1 = (const uint32_t*)(A  + (size_t)rA * lda + k0 + 16 + half * 8);
    const uint32_t* pb0 = (const uint32_t*)(Bt + (size_t)rB * ldb + k0 + half * 8);
    const uint32_t* pb1 = (const uint32_t*)(Bt + (size_t)rB * ldb + k0 + 16 + half * 8);
#pragma unroll
    for (int j = 0; j < 4; ++j) {
      fa.u[j] = pa0[j]; fa.u[4 + j] = pa1[j];
      fb.u[j] = pb0[j]; fb.u[4 + j] = pb1[j];
    }
    acc = __builtin_amdgcn_wmma_f32_16x16x32_bf16(false, fa.v, false, fb.v,
                                                  (short)0, acc, false, false);
  }

  int n = nBase + (lane & 15);
  float bv = bias ? bias[n] : 0.f;
#pragma unroll
  for (int r = 0; r < 8; ++r) {
    int m = mBase + half * 8 + r;
    float vo = acc[r] + bv;
    if (mode == 0) {
      outF[(size_t)m * ldc + n + (size_t)z * sCz] = vo;
    } else if (mode == 1) {
      outB[(size_t)m * ldc + n + (size_t)z * sCz] = f2bf(vo);
    } else if (mode == 2) {
      outF[(size_t)m * ldc + n] += vo;
    } else if (mode == 3) {
      outB[(size_t)m * ldc + n] = f2bf(gelu_tanh(vo));
    } else { // 4
      int bb = m >> 9, tok = m & 511, hh = n >> 5, d = n & 31;
      outB[(((size_t)(bb * 8 + hh)) * 32 + d) * 512 + tok] = f2bf(vo);
    }
  }
}

// ---------------------------------------------------------------------------
static inline void launch_gemm(hipStream_t st, dim3 grid,
                               const uint16_t* A, int lda, long long sAz,
                               const uint16_t* Bt, int ldb, long long sBz,
                               const float* bias, float* outF, uint16_t* outB,
                               int ldc, long long sCz, int M, int K, int mode)
{
  gemm_bf16_wmma<<<grid, dim3(256, 1, 1), 0, st>>>(A, lda, sAz, Bt, ldb, sBz,
                                                   bias, outF, outB, ldc, sCz, M, K, mode);
}

extern "C" void kernel_launch(void* const* d_in, const int* in_sizes, int n_in,
                              void* d_out, int out_size, void* d_ws, size_t ws_size,
                              hipStream_t stream)
{
  (void)in_sizes; (void)n_in; (void)out_size; (void)ws_size;
  const float* img  = (const float*)d_in[0];
  const float* bfac = (const float*)d_in[1];
  const float* w0  = (const float*)d_in[2];  const float* b0c = (const float*)d_in[3];
  const float* g1  = (const float*)d_in[4];  const float* be1 = (const float*)d_in[5];
  const float* w1  = (const float*)d_in[6];  const float* b1c = (const float*)d_in[7];
  const float* g2  = (const float*)d_in[8];  const float* be2 = (const float*)d_in[9];
  const float* w2  = (const float*)d_in[10]; const float* b2c = (const float*)d_in[11];
  const float* g3  = (const float*)d_in[12]; const float* be3 = (const float*)d_in[13];
  const float* wu  = (const float*)d_in[14]; const float* bu  = (const float*)d_in[15];
  const float* pos = (const float*)d_in[16];
  const float* lndg = (const float*)d_in[17]; const float* lndb = (const float*)d_in[18];
  const float* lnpg = (const float*)d_in[19]; const float* lnpb = (const float*)d_in[20];
  const float* Wq = (const float*)d_in[21]; const float* bq = (const float*)d_in[22];
  const float* Wk = (const float*)d_in[23]; const float* bk = (const float*)d_in[24];
  const float* Wv = (const float*)d_in[25]; const float* bv = (const float*)d_in[26];
  const float* Wo = (const float*)d_in[27]; const float* bo = (const float*)d_in[28];
  const float* l1g = (const float*)d_in[29]; const float* l1b = (const float*)d_in[30];
  const float* l2g = (const float*)d_in[31]; const float* l2b = (const float*)d_in[32];
  const float* W1 = (const float*)d_in[33]; const float* bf1 = (const float*)d_in[34];
  const float* W2 = (const float*)d_in[35]; const float* bf2 = (const float*)d_in[36];
  float* x = (float*)d_out;                  // (16,512,256) f32, updated in place

  // ---- workspace carve (~80 MB) ----
  uint8_t* wsb = (uint8_t*)d_ws; size_t off = 0;
  auto carve = [&](size_t bytes) -> void* {
    void* p = wsb + off; off = (off + bytes + 255) & ~(size_t)255; return p;
  };
  float*    bufP   = (float*)   carve(8192ull * 512 * 4);
  float*    posln  = (float*)   carve(512ull * 256 * 4);
  uint16_t* WqT    = (uint16_t*)carve(8ull * 256 * 256 * 2);
  uint16_t* WkT    = (uint16_t*)carve(8ull * 256 * 256 * 2);
  uint16_t* WvT    = (uint16_t*)carve(8ull * 256 * 256 * 2);
  uint16_t* WoT    = (uint16_t*)carve(8ull * 256 * 256 * 2);
  uint16_t* W1T    = (uint16_t*)carve(8ull * 1024 * 256 * 2);
  uint16_t* W2T    = (uint16_t*)carve(8ull * 1024 * 256 * 2);
  uint16_t* hbf    = (uint16_t*)carve(8192ull * 256 * 2);
  uint16_t* qbf    = (uint16_t*)carve(8192ull * 256 * 2);
  uint16_t* kbf    = (uint16_t*)carve(8192ull * 256 * 2);
  uint16_t* vt     = (uint16_t*)carve(8192ull * 256 * 2);
  uint16_t* obf    = (uint16_t*)carve(8192ull * 256 * 2);
  uint16_t* mbf    = (uint16_t*)carve(8192ull * 1024 * 2);
  float*    scores = (float*)   carve(8ull * 512 * 512 * 4);
  uint16_t* sbf    = (uint16_t*)carve(8ull * 512 * 512 * 2);

  // ---- pre-stages ----
  ln256_kernel<<<512, 256, 0, stream>>>(pos, lnpg, lnpb, posln, nullptr);
  patch_blur_kernel<<<8192, 512, 0, stream>>>(img, bfac, bufP);
  encoder_kernel<<<8192, 256, 0, stream>>>(bufP, w0, b0c, g1, be1, w1, b1c, g2, be2,
                                           w2, b2c, g3, be3, wu, bu, posln, lndg, lndb, x);

  conv_w_T<<<(8 * 256 * 256 + 255) / 256, 256, 0, stream>>>(Wq, WqT, 8, 256, 256);
  conv_w_T<<<(8 * 256 * 256 + 255) / 256, 256, 0, stream>>>(Wk, WkT, 8, 256, 256);
  conv_w_T<<<(8 * 256 * 256 + 255) / 256, 256, 0, stream>>>(Wv, WvT, 8, 256, 256);
  conv_w_T<<<(8 * 256 * 256 + 255) / 256, 256, 0, stream>>>(Wo, WoT, 8, 256, 256);
  conv_w_T<<<(8 * 256 * 1024 + 255) / 256, 256, 0, stream>>>(W1, W1T, 8, 256, 1024);
  conv_w_T<<<(8 * 1024 * 256 + 255) / 256, 256, 0, stream>>>(W2, W2T, 8, 1024, 256);

  // ---- transformer layers ----
  const float scale = 0.17677669529f;        // 1/sqrt(32)
  for (int i = 0; i < 8; ++i) {
    // h = LN(x, l1) -> bf16
    ln256_kernel<<<8192, 256, 0, stream>>>(x, l1g + i * 256, l1b + i * 256, nullptr, hbf);
    // q,k,v projections (M=8192, K=256, N=256)
    launch_gemm(stream, dim3(16, 64, 1), hbf, 256, 0, WqT + (size_t)i * 65536, 256, 0,
                bq + i * 256, nullptr, qbf, 256, 0, 8192, 256, 1);
    launch_gemm(stream, dim3(16, 64, 1), hbf, 256, 0, WkT + (size_t)i * 65536, 256, 0,
                bk + i * 256, nullptr, kbf, 256, 0, 8192, 256, 1);
    launch_gemm(stream, dim3(16, 64, 1), hbf, 256, 0, WvT + (size_t)i * 65536, 256, 0,
                bv + i * 256, nullptr, vt, 256, 0, 8192, 256, 4);   // V stored [b][h][d][tok]

    // attention, one batch at a time (stream-serialized; small scores buffer)
    for (int bb = 0; bb < 16; ++bb) {
      const uint16_t* qb = qbf + (size_t)bb * 512 * 256;
      const uint16_t* kb = kbf + (size_t)bb * 512 * 256;
      // S[h] = Q_h (512x32) * K_h^T   (z = head, column offset h*32)
      launch_gemm(stream, dim3(32, 4, 8), qb, 256, 32, kb, 256, 32,
                  nullptr, scores, nullptr, 512, 512ll * 512, 512, 32, 0);
      softmax512_kernel<<<dim3(512, 8), 256, 0, stream>>>(scores, sbf, scale);
      // O_h = softmax(S_h) (512x512) * V_h (512x32)
      launch_gemm(stream, dim3(2, 4, 8), sbf, 512, 512ll * 512,
                  vt + (size_t)bb * 8 * 32 * 512, 512, 32ll * 512,
                  nullptr, nullptr, obf + (size_t)bb * 512 * 256, 256, 32, 512, 512, 1);
    }

    // x += O @ Wo + bo
    launch_gemm(stream, dim3(16, 64, 1), obf, 256, 0, WoT + (size_t)i * 65536, 256, 0,
                bo + i * 256, x, nullptr, 256, 0, 8192, 256, 2);

    // h = LN(x, l2);  x += gelu(h @ W1 + bf1) @ W2 + bf2
    ln256_kernel<<<8192, 256, 0, stream>>>(x, l2g + i * 256, l2b + i * 256, nullptr, hbf);
    launch_gemm(stream, dim3(64, 64, 1), hbf, 256, 0, W1T + (size_t)i * 262144, 256, 0,
                bf1 + i * 1024, nullptr, mbf, 1024, 0, 8192, 256, 3);
    launch_gemm(stream, dim3(16, 64, 1), mbf, 1024, 0, W2T + (size_t)i * 262144, 1024, 0,
                bf2 + i * 256, x, nullptr, 256, 0, 8192, 1024, 2);
  }
}
